// GraphDAE_89283780149927
// MI455X (gfx1250) — matrix operand order repaired and checked by reading, not compiled, not run on hardware
//
#include <hip/hip_runtime.h>
#include <math.h>

// ---------------------------------------------------------------------------
// GraphDAE forward on gfx1250 (MI455X).
// GEMMs via v_wmma_f32_16x16x32_f16 (f32->f16 convert in LDS staging, f32 acc).
// Fragment loads are 2x ds_load_b128 each (LDS layouts match WMMA lane maps).
// GEMM is templated on (N, K, RELU): guards are constexpr-elided for regular
// shapes and branchless (clamped address + select) for ragged K=126 / N=140,
// so staging loads issue unconditionally and batch (no per-load waits).
// Small/irregular stages (D=10 encoder, K=2 convs, loss reductions) on VALU.
// Deterministic reductions: per-block partials + single finalize kernel.
// ---------------------------------------------------------------------------

typedef __attribute__((ext_vector_type(16))) _Float16 v16h;
typedef __attribute__((ext_vector_type(8)))  _Float16 h8v;
typedef __attribute__((ext_vector_type(8)))  float    v8f;

union AFrag { v16h v; h8v q[2]; };
union CFrag { v8f  v; float f[8]; };
union H2    { _Float16 h[2]; unsigned int u; };

static constexpr int BB   = 64;    // batch
static constexpr int NN   = 512;   // nodes
static constexpr int DD   = 10;    // node dim
static constexpr int HH   = 256;   // hidden
static constexpr int LL   = 128;   // latent
static constexpr int AAc  = 20;
static constexpr int SSc  = 7;
static constexpr int ATOM = 5;
static constexpr int ROWS = BB * NN;          // 32768
static constexpr int CLS  = AAc * SSc;        // 140
static constexpr int T1L  = LL - 1;           // 127 (conv1 out len)
static constexpr int T2L  = LL - 2;           // 126 (conv2 out len)

// ---------------------------------------------------------------------------
// Block-wide sum (blockDim.x == 256). Valid result on thread 0 only.
// ---------------------------------------------------------------------------
__device__ __forceinline__ float blockReduceSum256(float v) {
  __shared__ float sm[8];
  int lane = threadIdx.x & 31, wid = threadIdx.x >> 5;
#pragma unroll
  for (int off = 16; off > 0; off >>= 1) v += __shfl_down(v, off, 32);
  __syncthreads();
  if (lane == 0) sm[wid] = v;
  __syncthreads();
  float r = 0.f;
  if (threadIdx.x == 0) {
#pragma unroll
    for (int k = 0; k < 8; ++k) r += sm[k];
  }
  return r;
}

// ---------------------------------------------------------------------------
// Generic GEMM: C[M,N] = op(A[M,K] @ B[K,N] + bias[N]),  op = ReLU optional.
// fp32 in memory, f16 WMMA compute, fp32 accumulate.
// Tile BM=64 x BN=128 x BK=32, 256 threads = 8 waves (4 along M, 2 along N).
// M (grid.y * 64) must divide evenly: all call sites use M = 32768.
// sA: row-major [64][32] halves -> A frag = 2x ds_load_b128.
// sBt: transposed [128][40] halves (K contiguous) -> B frag = 2x ds_load_b128.
// ---------------------------------------------------------------------------
template <int N, int K, bool RELU>
__global__ __launch_bounds__(256) void gemm_wmma(
    const float* __restrict__ A, const float* __restrict__ Bw,
    const float* __restrict__ bias, float* __restrict__ C) {
  constexpr int BM = 64, BN = 128, BK = 32;
  constexpr bool RAGK = (K % BK) != 0;  // K even at all call sites
  constexpr bool RAGN = (N % BN) != 0;
  constexpr int NK = (K + BK - 1) / BK;
  __shared__ alignas(16) _Float16 sA[BM][BK];        // 4 KB, row = 64 B
  __shared__ alignas(16) _Float16 sBt[BN][BK + 8];   // 10 KB, row = 80 B

  const int tid  = threadIdx.x;
  const int lane = tid & 31;
  const int wave = tid >> 5;
  const int wm   = wave & 3;   // 0..3 -> 16-row subtile in M
  const int wn   = wave >> 2;  // 0..1 -> 64-col half in N

  const int blockM = blockIdx.y * BM;
  const int blockN = blockIdx.x * BN;

  CFrag acc[4];
#pragma unroll
  for (int j = 0; j < 4; ++j)
#pragma unroll
    for (int i = 0; i < 8; ++i) acc[j].f[i] = 0.f;

#pragma unroll 1
  for (int kt = 0; kt < NK; ++kt) {
    const int kk = kt * BK;
    // stage A tile (64x32): float2 load (8B-aligned: K,gk even), pack 2xf16
#pragma unroll
    for (int s = 0; s < (BM * BK) / 512; ++s) {  // 4 pairs per thread
      const int p = tid + s * 256;               // pair index, 16 pairs/row
      const int r = p >> 4;
      const int k2 = (p & 15) * 2;
      const int gk = kk + k2;
      const float* ar = A + (size_t)(blockM + r) * K;
      float v0, v1;
      if constexpr (!RAGK) {
        const float2 t2 = *(const float2*)(ar + gk);
        v0 = t2.x; v1 = t2.y;
      } else {
        const bool ok = (gk + 1) < K;            // pair-granular (K even)
        const float2 t2 = *(const float2*)(ar + (ok ? gk : 0));
        v0 = ok ? t2.x : 0.f;
        v1 = ok ? t2.y : 0.f;
      }
      H2 t; t.h[0] = (_Float16)v0; t.h[1] = (_Float16)v1;
      *(unsigned int*)&sA[r][k2] = t.u;
    }
    // stage B tile (32x128) transposed: coalesced along N, packed K pairs
#pragma unroll
    for (int s = 0; s < (BK * BN) / 512; ++s) {  // 8 pairs per thread
      const int p = tid + s * 256;               // 0..2047
      const int k2 = (p >> 7) * 2;               // 0,2,..,30
      const int c = p & 127;
      const int gn = blockN + c;
      const int gk = kk + k2;
      float v0, v1;
      if constexpr (!RAGK && !RAGN) {
        v0 = Bw[(size_t)gk * N + gn];
        v1 = Bw[(size_t)(gk + 1) * N + gn];
      } else {
        const bool okn = RAGN ? (gn < N) : true;
        const bool okk = RAGK ? ((gk + 1) < K) : true;
        const int gnc = okn ? gn : (N - 1);
        const int gkc = okk ? gk : 0;
        v0 = Bw[(size_t)gkc * N + gnc];
        v1 = Bw[(size_t)(gkc + 1) * N + gnc];
        if (!(okn && okk)) { v0 = 0.f; v1 = 0.f; }
      }
      H2 t; t.h[0] = (_Float16)v0; t.h[1] = (_Float16)v1;
      *(unsigned int*)&sBt[c][k2] = t.u;
    }
    __syncthreads();

    // A fragment: lane l row (l&15); K(e) = (e&8 ? e+8 : e) + (l>>4)*8
    // -> chunks [kadd, kadd+8) and [kadd+16, kadd+24): 2x b128
    AFrag af;
    const int ar = wm * 16 + (lane & 15);
    const int kadd = (lane >> 4) * 8;
    af.q[0] = *(const h8v*)&sA[ar][kadd];
    af.q[1] = *(const h8v*)&sA[ar][kadd + 16];

    // B fragments: lane l col (l&15); K(e) = (e&8 ? e+8 : e) + (l>>4)*8
    const int bkadd = (lane >> 4) * 8;
#pragma unroll
    for (int j = 0; j < 4; ++j) {
      AFrag bf;
      const int bc = wn * 64 + j * 16 + (lane & 15);
      bf.q[0] = *(const h8v*)&sBt[bc][bkadd];
      bf.q[1] = *(const h8v*)&sBt[bc][bkadd + 16];
      acc[j].v = __builtin_amdgcn_wmma_f32_16x16x32_f16(
          false, af.v, false, bf.v, (short)0, acc[j].v, false, false);
    }
    __syncthreads();
  }

  // epilogue: D row = i + (l>=16 ? 8 : 0), col = l&15
  const int mrow0 = blockM + wm * 16 + ((lane >> 4) << 3);
#pragma unroll
  for (int j = 0; j < 4; ++j) {
    const int n = blockN + wn * 64 + j * 16 + (lane & 15);
    if (!RAGN || n < N) {
      const float bv = bias[n];
#pragma unroll
      for (int i = 0; i < 8; ++i) {
        float v = acc[j].f[i] + bv;
        if (RELU) v = fmaxf(v, 0.f);
        C[(size_t)(mrow0 + i) * N + n] = v;
      }
    }
  }
}

// ---------------------------------------------------------------------------
// Encoder: h = relu(mask * ((adj@x)/deg @ W_rel + b_rel + x @ W_root))
// One wave per node row; lanes stride neighbors j, then fan out over H.
// ---------------------------------------------------------------------------
__global__ __launch_bounds__(256) void encoder_kernel(
    const float* __restrict__ x, const float* __restrict__ adj,
    const unsigned char* __restrict__ mask, const float* __restrict__ W_rel,
    const float* __restrict__ b_rel, const float* __restrict__ W_root,
    float* __restrict__ h) {
  const int tid = threadIdx.x, wave = tid >> 5, lane = tid & 31;
  const int row = blockIdx.x * 8 + wave;  // b*512 + i
  const int b = row >> 9;
  const float* adjr = adj + (size_t)row * NN;

  float acc[DD];
#pragma unroll
  for (int d = 0; d < DD; ++d) acc[d] = 0.f;
  float deg = 0.f;
  for (int t = 0; t < NN / 32; ++t) {
    const int j = lane + t * 32;
    const float a = adjr[j];
    const float* xr = x + ((size_t)(b << 9) + j) * DD;
    deg += a;
#pragma unroll
    for (int d = 0; d < DD; ++d) acc[d] += a * xr[d];
  }
#pragma unroll
  for (int off = 16; off > 0; off >>= 1) {
    deg += __shfl_down(deg, off, 32);
#pragma unroll
    for (int d = 0; d < DD; ++d) acc[d] += __shfl_down(acc[d], off, 32);
  }
  deg = __shfl(deg, 0, 32);
#pragma unroll
  for (int d = 0; d < DD; ++d) acc[d] = __shfl(acc[d], 0, 32);
  const float dinv = 1.f / fmaxf(deg, 1.f);

  float xi[DD];
  const float* xrow = x + (size_t)row * DD;
#pragma unroll
  for (int d = 0; d < DD; ++d) xi[d] = xrow[d];
  const bool m = mask[row] != 0;

  const int c0 = lane * 8;
#pragma unroll
  for (int u = 0; u < 8; ++u) {
    const int c = c0 + u;
    float s = b_rel[c];
#pragma unroll
    for (int d = 0; d < DD; ++d)
      s += (acc[d] * dinv) * W_rel[d * HH + c] + xi[d] * W_root[d * HH + c];
    h[(size_t)row * HH + c] = m ? fmaxf(s, 0.f) : 0.f;
  }
}

// ---------------------------------------------------------------------------
// z = mu + eps * exp(0.5*lv); KL partials: sum(1 + lv - mu^2 - exp(lv))
// ---------------------------------------------------------------------------
__global__ __launch_bounds__(256) void zkl_kernel(
    const float* __restrict__ mu, const float* __restrict__ lv,
    const float* __restrict__ eps, float* __restrict__ z,
    float* __restrict__ kl_part) {
  const size_t total = (size_t)ROWS * LL;
  const size_t stride = (size_t)gridDim.x * blockDim.x;
  float acc = 0.f;
  for (size_t i = (size_t)blockIdx.x * blockDim.x + threadIdx.x; i < total;
       i += stride) {
    const float m = mu[i], l = lv[i];
    z[i] = m + eps[i] * expf(0.5f * l);
    acc += 1.f + l - m * m - expf(l);
  }
  const float s = blockReduceSum256(acc);
  if (threadIdx.x == 0) kl_part[blockIdx.x] = s;
}

// ---------------------------------------------------------------------------
// out_x_atom = ha @ W_fca + b_fca  (N=5, VALU) fused with MSE partials.
// ---------------------------------------------------------------------------
__global__ __launch_bounds__(256) void atom_kernel(
    const float* __restrict__ ha, const float* __restrict__ W_fca,
    const float* __restrict__ b_fca, const float* __restrict__ x,
    float* __restrict__ out_atom, float* __restrict__ mse_part) {
  const int gid = blockIdx.x * 256 + threadIdx.x;  // 0..163839 exactly
  const int row = gid / ATOM, c = gid % ATOM;
  const float* har = ha + (size_t)row * HH;
  float s = b_fca[c];
  for (int k = 0; k < HH; ++k) s += har[k] * W_fca[k * ATOM + c];
  out_atom[gid] = s;
  const float d = s - x[(size_t)row * DD + ATOM + c];
  const float r = blockReduceSum256(d * d);
  if (threadIdx.x == 0) mse_part[blockIdx.x] = r;
}

// ---------------------------------------------------------------------------
// Conv1d 'VALID', K=2: c1[b,o,t] = relu(sum_c z[b,c,t..t+1] . W[o,c,:] + b[o])
// ---------------------------------------------------------------------------
__global__ __launch_bounds__(128) void conv1_kernel(
    const float* __restrict__ z, const float* __restrict__ Wc1,
    const float* __restrict__ bc1, float* __restrict__ c1) {
  const int t = threadIdx.x, o = blockIdx.x, b = blockIdx.y;
  if (t >= T1L) return;
  const float* zb = z + (size_t)b * NN * LL;
  const float* w = Wc1 + (size_t)o * NN * 2;
  float acc = 0.f;
  for (int c = 0; c < NN; ++c) {
    const float* zr = zb + (size_t)c * LL + t;
    acc = fmaf(zr[0], w[2 * c], acc);
    acc = fmaf(zr[1], w[2 * c + 1], acc);
  }
  c1[((size_t)b * 256 + o) * T1L + t] = fmaxf(acc + bc1[o], 0.f);
}

__global__ __launch_bounds__(128) void conv2_kernel(
    const float* __restrict__ c1, const float* __restrict__ Wc2,
    const float* __restrict__ bc2, float* __restrict__ c2) {
  const int t = threadIdx.x, o = blockIdx.x, b = blockIdx.y;
  if (t >= T2L) return;
  const float* cb = c1 + (size_t)b * 256 * T1L;
  const float* w = Wc2 + (size_t)o * 256 * 2;
  float acc = 0.f;
  for (int c = 0; c < 256; ++c) {
    const float* cr = cb + (size_t)c * T1L + t;
    acc = fmaf(cr[0], w[2 * c], acc);
    acc = fmaf(cr[1], w[2 * c + 1], acc);
  }
  c2[((size_t)b * NN + o) * T2L + t] = fmaxf(acc + bc2[o], 0.f);
}

// ---------------------------------------------------------------------------
// Cross-entropy partials: one wave per node row over 140 logits.
// ---------------------------------------------------------------------------
__global__ __launch_bounds__(256) void ce_kernel(
    const float* __restrict__ out_class, const float* __restrict__ x,
    float* __restrict__ ce_part) {
  __shared__ float sw[8];
  const int tid = threadIdx.x, wave = tid >> 5, lane = tid & 31;
  const int row = blockIdx.x * 8 + wave;
  const float* lg = out_class + (size_t)row * CLS;

  float v[5];
#pragma unroll
  for (int t = 0; t < 5; ++t) {
    const int e = lane + 32 * t;
    v[t] = (e < CLS) ? lg[e] : -3.0e38f;
  }
  float mx = v[0];
#pragma unroll
  for (int t = 1; t < 5; ++t) mx = fmaxf(mx, v[t]);
#pragma unroll
  for (int off = 16; off > 0; off >>= 1) mx = fmaxf(mx, __shfl_xor(mx, off, 32));
  float s = 0.f;
#pragma unroll
  for (int t = 0; t < 5; ++t) {
    const int e = lane + 32 * t;
    if (e < CLS) s += expf(v[t] - mx);
  }
#pragma unroll
  for (int off = 16; off > 0; off >>= 1) s += __shfl_xor(s, off, 32);

  if (lane == 0) {
    const float xa = x[(size_t)row * DD + 0];
    const float xs = x[(size_t)row * DD + 1];
    const int map = (int)((xs - 1.f) * (float)AAc + (xa - 1.f));
    const int mc = map < 0 ? 0 : map;
    sw[wave] = (map >= 0) ? (mx + logf(s) - lg[mc]) : 0.f;
  }
  __syncthreads();
  if (tid == 0) {
    float r = 0.f;
#pragma unroll
    for (int k = 0; k < 8; ++k) r += sw[k];
    ce_part[blockIdx.x] = r;
  }
}

// ---------------------------------------------------------------------------
// BCE-with-logits partials over out_adj vs binary adjacency.
// ---------------------------------------------------------------------------
__global__ __launch_bounds__(256) void bce_kernel(
    const float* __restrict__ out_adj, const float* __restrict__ adj,
    const unsigned char* __restrict__ mask, float* __restrict__ bce_part) {
  const size_t total = (size_t)BB * NN * NN;
  const size_t stride = (size_t)gridDim.x * blockDim.x;
  float acc = 0.f;
  for (size_t idx = (size_t)blockIdx.x * blockDim.x + threadIdx.x; idx < total;
       idx += stride) {
    const int j = (int)(idx & (NN - 1));
    const size_t ri = idx >> 9;  // b*512 + i
    const int i = (int)(ri & (NN - 1));
    const int b = (int)(ri >> 9);
    const float l = out_adj[idx];
    float t = 0.f;
    if (mask[ri] && mask[((size_t)b << 9) + j]) {
      if (i == j || adj[idx] > 0.f) t = 1.f;
    }
    acc += fmaxf(l, 0.f) - l * t + log1pf(expf(-fabsf(l)));
  }
  const float r = blockReduceSum256(acc);
  if (threadIdx.x == 0) bce_part[blockIdx.x] = r;
}

// ---------------------------------------------------------------------------
// Final scalar loss.
// ---------------------------------------------------------------------------
__global__ __launch_bounds__(256) void finalize_kernel(
    const float* __restrict__ ce_p, const float* __restrict__ bce_p,
    const float* __restrict__ mse_p, const float* __restrict__ kl_p,
    float* __restrict__ out) {
  const int tid = threadIdx.x;
  float a = 0.f;
  for (int i = tid; i < 4096; i += 256) a += ce_p[i];
  const float ceS = blockReduceSum256(a);
  a = 0.f;
  for (int i = tid; i < 1024; i += 256) a += bce_p[i];
  const float bceS = blockReduceSum256(a);
  a = 0.f;
  for (int i = tid; i < 640; i += 256) a += mse_p[i];
  const float mseS = blockReduceSum256(a);
  a = 0.f;
  for (int i = tid; i < 256; i += 256) a += kl_p[i];
  const float klS = blockReduceSum256(a);
  if (tid == 0) {
    const float ce = ceS / (float)ROWS;
    const float bce = bceS / (float)((size_t)BB * NN * NN);
    const float mse = mseS / (float)(ROWS * ATOM);
    const float kl = -0.5f * klS;
    const float beta = fabsf((ce + bce) / mse);
    out[0] = ce + beta * mse + bce + 1e-3f * kl;
  }
}

// ---------------------------------------------------------------------------
extern "C" void kernel_launch(void* const* d_in, const int* in_sizes, int n_in,
                              void* d_out, int out_size, void* d_ws,
                              size_t ws_size, hipStream_t stream) {
  (void)in_sizes; (void)n_in; (void)out_size; (void)ws_size;
  const float* x      = (const float*)d_in[0];
  const float* adj    = (const float*)d_in[1];
  const unsigned char* mask = (const unsigned char*)d_in[2];
  const float* eps    = (const float*)d_in[3];
  const float* W_rel  = (const float*)d_in[4];
  const float* b_rel  = (const float*)d_in[5];
  const float* W_root = (const float*)d_in[6];
  const float* W_mu   = (const float*)d_in[7];
  const float* b_mu   = (const float*)d_in[8];
  const float* W_lv   = (const float*)d_in[9];
  const float* b_lv   = (const float*)d_in[10];
  const float* Wx1 = (const float*)d_in[11]; const float* bx1 = (const float*)d_in[12];
  const float* Wx2 = (const float*)d_in[13]; const float* bx2 = (const float*)d_in[14];
  const float* Wx3 = (const float*)d_in[15]; const float* bx3 = (const float*)d_in[16];
  const float* Wa1 = (const float*)d_in[17]; const float* ba1 = (const float*)d_in[18];
  const float* Wa2 = (const float*)d_in[19]; const float* ba2 = (const float*)d_in[20];
  const float* Wa3 = (const float*)d_in[21]; const float* ba3 = (const float*)d_in[22];
  const float* Wc1 = (const float*)d_in[23]; const float* bc1 = (const float*)d_in[24];
  const float* Wc2 = (const float*)d_in[25]; const float* bc2 = (const float*)d_in[26];
  const float* W_fcx = (const float*)d_in[27]; const float* b_fcx = (const float*)d_in[28];
  const float* W_fca = (const float*)d_in[29]; const float* b_fca = (const float*)d_in[30];
  const float* W_fcj = (const float*)d_in[31]; const float* b_fcj = (const float*)d_in[32];

  // workspace layout (floats); mu/lv regions are reused for c1/c2 after z.
  float* ws = (float*)d_ws;
  const size_t o_h  = 0;
  const size_t o_mu = o_h  + (size_t)ROWS * HH;   // 8.39M
  const size_t o_lv = o_mu + (size_t)ROWS * LL;   // 4.19M
  const size_t o_z  = o_lv + (size_t)ROWS * LL;
  const size_t o_t1 = o_z  + (size_t)ROWS * LL;
  const size_t o_t2 = o_t1 + (size_t)ROWS * 512;
  const size_t o_hx = o_t2 + (size_t)ROWS * 512;
  const size_t o_ha = o_hx + (size_t)ROWS * HH;
  const size_t o_ce = o_ha + (size_t)ROWS * HH;
  const size_t o_bc = o_ce + 4096;
  const size_t o_ms = o_bc + 1024;
  const size_t o_kl = o_ms + 640;
  float *h = ws + o_h, *mu = ws + o_mu, *lv = ws + o_lv, *z = ws + o_z;
  float *t1 = ws + o_t1, *t2 = ws + o_t2, *hx = ws + o_hx, *ha = ws + o_ha;
  float *c1 = ws + o_mu, *c2 = ws + o_lv;  // reuse: mu/lv dead after zkl
  float *pce = ws + o_ce, *pbce = ws + o_bc, *pmse = ws + o_ms, *pkl = ws + o_kl;

  float* out = (float*)d_out;
  float* out_class = out + 1;
  float* out_atom  = out_class + (size_t)ROWS * CLS;
  float* out_adjp  = out_atom + (size_t)ROWS * ATOM;

  // --- encoder ---
  encoder_kernel<<<ROWS / 8, 256, 0, stream>>>(x, adj, mask, W_rel, b_rel, W_root, h);
  gemm_wmma<LL, HH, false><<<dim3(1, ROWS / 64), 256, 0, stream>>>(h, W_mu, b_mu, mu);
  gemm_wmma<LL, HH, false><<<dim3(1, ROWS / 64), 256, 0, stream>>>(h, W_lv, b_lv, lv);
  zkl_kernel<<<256, 256, 0, stream>>>(mu, lv, eps, z, pkl);

  // --- gen_x MLP: z->512->512->256 -> class head ---
  gemm_wmma<512, LL, true><<<dim3(4, ROWS / 64), 256, 0, stream>>>(z, Wx1, bx1, t1);
  gemm_wmma<512, 512, true><<<dim3(4, ROWS / 64), 256, 0, stream>>>(t1, Wx2, bx2, t2);
  gemm_wmma<HH, 512, false><<<dim3(2, ROWS / 64), 256, 0, stream>>>(t2, Wx3, bx3, hx);
  gemm_wmma<CLS, HH, false><<<dim3(2, ROWS / 64), 256, 0, stream>>>(hx, W_fcx, b_fcx, out_class);

  // --- gen_x_atomic MLP -> atom head (+MSE partials) ---
  gemm_wmma<512, LL, true><<<dim3(4, ROWS / 64), 256, 0, stream>>>(z, Wa1, ba1, t1);
  gemm_wmma<512, 512, true><<<dim3(4, ROWS / 64), 256, 0, stream>>>(t1, Wa2, ba2, t2);
  gemm_wmma<HH, 512, false><<<dim3(2, ROWS / 64), 256, 0, stream>>>(t2, Wa3, ba3, ha);
  atom_kernel<<<(ROWS * ATOM) / 256, 256, 0, stream>>>(ha, W_fca, b_fca, x, out_atom, pmse);

  // --- conv stack -> adjacency head ---
  conv1_kernel<<<dim3(256, BB), 128, 0, stream>>>(z, Wc1, bc1, c1);
  conv2_kernel<<<dim3(512, BB), 128, 0, stream>>>(c1, Wc2, bc2, c2);
  gemm_wmma<NN, T2L, false><<<dim3(4, ROWS / 64), 256, 0, stream>>>(c2, W_fcj, b_fcj, out_adjp);

  // --- losses ---
  ce_kernel<<<ROWS / 8, 256, 0, stream>>>(out_class, x, pce);
  bce_kernel<<<1024, 256, 0, stream>>>(out_adjp, adj, mask, pbce);
  finalize_kernel<<<1, 256, 0, stream>>>(pce, pbce, pmse, pkl, out);
}
